// Fusion_77163382440918
// MI455X (gfx1250) — matrix operand order repaired and checked by reading, not compiled, not run on hardware
//
#include <hip/hip_runtime.h>
#include <cstdint>
#include <cstddef>

#define DCH 128
#define NCPT 128
#define NITEM 20000
#define NUSER 50000
#define EK_E 2048
#define EEK_E 100000
#define EUE_E 1000000

typedef _Float16 h8_t  __attribute__((ext_vector_type(8)));
typedef _Float16 h16_t __attribute__((ext_vector_type(16)));
typedef float    f8_t  __attribute__((ext_vector_type(8)));

// ---------- helpers ----------
__device__ __forceinline__ unsigned fenc(float f) {
    unsigned u = __float_as_uint(f);
    return (u & 0x80000000u) ? ~u : (u | 0x80000000u);
}
__device__ __forceinline__ float fdec(unsigned u) {
    return (u & 0x80000000u) ? __uint_as_float(u & 0x7FFFFFFFu) : __uint_as_float(~u);
}
__device__ __forceinline__ float dot4(float4 a, float4 b) {
    return a.x * b.x + a.y * b.y + a.z * b.z + a.w * b.w;
}

// ---------- fill ----------
__global__ void fill_u32(unsigned* __restrict__ p, unsigned v, int n) {
    int i = blockIdx.x * blockDim.x + threadIdx.x;
    if (i < n) p[i] = v;
}

// ---------- f32 -> f16 convert ----------
__global__ void cvt_f16(const float* __restrict__ in, _Float16* __restrict__ out, int n) {
    int i = blockIdx.x * blockDim.x + threadIdx.x;
    if (i < n) out[i] = (_Float16)in[i];
}

// ---------- WMMA GEMM: Z[M,128] = H16[M,128] @ W16[128,128]^T  (f32 accum) ----------
// Each wave owns one 16-row tile of H: the full K=128 A-slice is held in registers
// (read from HBM exactly once) and reused across all 8 column tiles. B (the 128x128
// weight matrix) is L2-resident. Two independent accumulator chains per tile hide
// the WMMA->WMMA D->C hazard latency.
__global__ void wmma_gemm(const _Float16* __restrict__ H, const _Float16* __restrict__ W,
                          float* __restrict__ Z, int M) {
    const int wave = threadIdx.x >> 5;     // 8 waves: 8 row tiles per block
    const int lane = threadIdx.x & 31;
    const int lane16 = lane & 15;
    const int hsel = lane >> 4;            // 0|1 selects K-half per ISA layout
    const int m0 = (blockIdx.x * 8 + wave) * 16;
    if (m0 >= M) return;                   // wave-uniform

    int arow = m0 + lane16;
    if (arow >= M) arow = M - 1;           // clamp (tail tiles only)
    const _Float16* hrow = H + (size_t)arow * DCH;

    // A (16x32 f16) per k-step: lanes 0-15: K = k0+0..7, k0+16..23 ; lanes 16-31: +8
    h16_t A[4];
#pragma unroll
    for (int i = 0; i < 4; ++i) {
        const int k0 = i * 32;
        h8_t a_lo = *(const h8_t*)(hrow + k0 + hsel * 8);
        h8_t a_hi = *(const h8_t*)(hrow + k0 + 16 + hsel * 8);
        A[i] = __builtin_shufflevector(a_lo, a_hi, 0,1,2,3,4,5,6,7,8,9,10,11,12,13,14,15);
    }

    const bool full = (m0 + 16 <= M);
#pragma unroll
    for (int nt = 0; nt < 8; ++nt) {
        const int n0 = nt * 16;
        const _Float16* wrow = W + (size_t)(n0 + lane16) * DCH;
        f8_t acc0 = {}, acc1 = {};
#pragma unroll
        for (int i = 0; i < 4; i += 2) {
            // B (32x16 f16): lane column n = lane16; lanes 0-15 hold K=k0+0..7, k0+16..23,
            // lanes 16-31 hold K=k0+8..15, k0+24..31 (same order as A) from W row n.
            const int ka = i * 32, kb = (i + 1) * 32;
            h8_t b0l = *(const h8_t*)(wrow + ka + hsel * 8);
            h8_t b0h = *(const h8_t*)(wrow + ka + 16 + hsel * 8);
            h8_t b1l = *(const h8_t*)(wrow + kb + hsel * 8);
            h8_t b1h = *(const h8_t*)(wrow + kb + 16 + hsel * 8);
            h16_t B0 = __builtin_shufflevector(b0l, b0h, 0,1,2,3,4,5,6,7,8,9,10,11,12,13,14,15);
            h16_t B1 = __builtin_shufflevector(b1l, b1h, 0,1,2,3,4,5,6,7,8,9,10,11,12,13,14,15);
            acc0 = __builtin_amdgcn_wmma_f32_16x16x32_f16(false, A[i], false, B0, (short)0,
                                                          acc0, false, false);
            acc1 = __builtin_amdgcn_wmma_f32_16x16x32_f16(false, A[i + 1], false, B1, (short)0,
                                                          acc1, false, false);
        }
        f8_t acc = acc0 + acc1;

        // C/D layout: VGPR r, lane -> N = lane16, M = r + hsel*8
        const int col = n0 + lane16;
        float* zp = Z + (size_t)m0 * DCH + col;
        if (full) {
#pragma unroll
            for (int r = 0; r < 8; ++r)
                zp[(size_t)(r + hsel * 8) * DCH] = acc[r];
        } else {
#pragma unroll
            for (int r = 0; r < 8; ++r) {
                int row = m0 + r + hsel * 8;
                if (row < M) Z[(size_t)row * DCH + col] = acc[r];
            }
        }
    }
}

// ---------- per-node attention scalars: es[n]=z[n].a[:128], ed[n]=z[n].a[128:] ----------
__global__ void node_scalars(const float* __restrict__ Z, const float* __restrict__ attn,
                             float* __restrict__ es, float* __restrict__ ed, int n) {
    int wave = threadIdx.x >> 5, lane = threadIdx.x & 31;
    int row = blockIdx.x * (blockDim.x >> 5) + wave;
    if (row >= n) return;
    float4 z4 = ((const float4*)(Z + (size_t)row * DCH))[lane];
    float4 a1 = ((const float4*)attn)[lane];
    float4 a2 = ((const float4*)(attn + DCH))[lane];
    float s = dot4(z4, a1);
    float t = dot4(z4, a2);
#pragma unroll
    for (int off = 16; off; off >>= 1) {
        s += __shfl_xor(s, off, 32);
        t += __shfl_xor(t, off, 32);
    }
    if (lane == 0) { es[row] = s; ed[row] = t; }
}

// ---------- edge pass 1: e = es[src] + ed[dst]; segment max (monotone-uint atomicMax) ----
__global__ void edge_logits(const int* __restrict__ src, const int* __restrict__ dst,
                            const float* __restrict__ es, const float* __restrict__ ed,
                            float* __restrict__ ew, unsigned* __restrict__ menc, int nE) {
    int i = blockIdx.x * blockDim.x + threadIdx.x;
    if (i >= nE) return;
    float e = es[src[i]] + ed[dst[i]];
    ew[i] = e;
    atomicMax(&menc[dst[i]], fenc(e));
}

// ---------- edge pass 2: w = exp(e - m[dst]); segment sum ----------
__global__ void edge_expw(const int* __restrict__ dst, const unsigned* __restrict__ menc,
                          float* __restrict__ ew, float* __restrict__ denom, int nE) {
    int i = blockIdx.x * blockDim.x + threadIdx.x;
    if (i >= nE) return;
    int d = dst[i];
    float w = __expf(ew[i] - fdec(menc[d]));
    ew[i] = w;
    atomicAdd(&denom[d], w);
}

// ---------- edge pass 3: agg[dst-base] += (w/denom[dst]) * z[src]  (wave per edge) ------
__global__ void edge_scatter(const int* __restrict__ src, const int* __restrict__ dst,
                             const float* __restrict__ ew, const float* __restrict__ denom,
                             const float* __restrict__ Z, float* __restrict__ agg,
                             int dstBase, int dstCount, int nE) {
    int gid = blockIdx.x * blockDim.x + threadIdx.x;
    int e = gid >> 5, lane = gid & 31;
    if (e >= nE) return;
    int dn = dst[e];
    int d = dn - dstBase;
    if (d < 0 || d >= dstCount) return;
    float alpha = ew[e] / denom[dn];
    float4 z4 = ((const float4*)(Z + (size_t)src[e] * DCH))[lane];
    float* ag = agg + (size_t)d * DCH + lane * 4;
    atomicAdd(ag + 0, alpha * z4.x);
    atomicAdd(ag + 1, alpha * z4.y);
    atomicAdd(ag + 2, alpha * z4.z);
    atomicAdd(ag + 3, alpha * z4.w);
}

// ---------- fusion heads ----------
__global__ void kn_fuse(const float* __restrict__ A, const float* __restrict__ Bm,
                        const float* __restrict__ Cm, const float* __restrict__ Dm,
                        const float* k1w, const float* k1b, const float* k2w, const float* k2b,
                        const float* k3w, const float* k3b, float* __restrict__ out) {
    int wave = threadIdx.x >> 5, lane = threadIdx.x & 31;
    int row = blockIdx.x * (blockDim.x >> 5) + wave;
    if (row >= NCPT) return;
    float4 a4 = ((const float4*)(A + (size_t)row * DCH))[lane];
    float4 b4 = ((const float4*)(Bm + (size_t)row * DCH))[lane];
    float4 c4 = ((const float4*)(Cm + (size_t)row * DCH))[lane];
    float4 d4 = ((const float4*)(Dm + (size_t)row * DCH))[lane];
    float s1 = dot4(a4, ((const float4*)k1w)[lane]) + dot4(b4, ((const float4*)(k1w + DCH))[lane]);
    float s2 = dot4(a4, ((const float4*)k2w)[lane]) + dot4(c4, ((const float4*)(k2w + DCH))[lane]);
    float s3 = dot4(a4, ((const float4*)k3w)[lane]) + dot4(d4, ((const float4*)(k3w + DCH))[lane]);
#pragma unroll
    for (int off = 16; off; off >>= 1) {
        s1 += __shfl_xor(s1, off, 32);
        s2 += __shfl_xor(s2, off, 32);
        s3 += __shfl_xor(s3, off, 32);
    }
    s1 += k1b[0]; s2 += k2b[0]; s3 += k3b[0];
    float mx = fmaxf(s1, fmaxf(s2, s3));
    float e1 = __expf(s1 - mx), e2 = __expf(s2 - mx), e3 = __expf(s3 - mx);
    float inv = 1.0f / (e1 + e2 + e3);
    float4 o;
    o.x = a4.x + (e1 * b4.x + e2 * c4.x + e3 * d4.x) * inv;
    o.y = a4.y + (e1 * b4.y + e2 * c4.y + e3 * d4.y) * inv;
    o.z = a4.z + (e1 * b4.z + e2 * c4.z + e3 * d4.z) * inv;
    o.w = a4.w + (e1 * b4.w + e2 * c4.w + e3 * d4.w) * inv;
    ((float4*)(out + (size_t)row * DCH))[lane] = o;
}

__global__ void exer_fuse(const float* __restrict__ A, const float* __restrict__ Bm,
                          const float* __restrict__ Cm,
                          const float* e1w, const float* e1b, const float* e2w, const float* e2b,
                          float* __restrict__ out) {
    int wave = threadIdx.x >> 5, lane = threadIdx.x & 31;
    int row = blockIdx.x * (blockDim.x >> 5) + wave;
    if (row >= NITEM) return;
    float4 a4 = ((const float4*)(A + (size_t)row * DCH))[lane];
    float4 b4 = ((const float4*)(Bm + (size_t)row * DCH))[lane];
    float4 c4 = ((const float4*)(Cm + (size_t)row * DCH))[lane];
    float t1 = dot4(a4, ((const float4*)e1w)[lane]) + dot4(b4, ((const float4*)(e1w + DCH))[lane]);
    float t2 = dot4(a4, ((const float4*)e2w)[lane]) + dot4(c4, ((const float4*)(e2w + DCH))[lane]);
#pragma unroll
    for (int off = 16; off; off >>= 1) {
        t1 += __shfl_xor(t1, off, 32);
        t2 += __shfl_xor(t2, off, 32);
    }
    t1 += e1b[0]; t2 += e2b[0];
    float mx = fmaxf(t1, t2);
    float x1 = __expf(t1 - mx), x2 = __expf(t2 - mx);
    float inv = 1.0f / (x1 + x2);
    float4 o;
    o.x = a4.x + (x1 * b4.x + x2 * c4.x) * inv;
    o.y = a4.y + (x1 * b4.y + x2 * c4.y) * inv;
    o.z = a4.z + (x1 * b4.z + x2 * c4.z) * inv;
    o.w = a4.w + (x1 * b4.w + x2 * c4.w) * inv;
    ((float4*)(out + (size_t)row * DCH))[lane] = o;
}

__global__ void stu_add(const float* __restrict__ stu, const float* __restrict__ agg,
                        float* __restrict__ out, int n) {
    int i = blockIdx.x * blockDim.x + threadIdx.x;
    if (i < n) out[i] = stu[i] + agg[i];
}

// ---------- host-side orchestration ----------
extern "C" void kernel_launch(void* const* d_in, const int* in_sizes, int n_in,
                              void* d_out, int out_size, void* d_ws, size_t ws_size,
                              hipStream_t stream) {
    (void)in_sizes; (void)n_in; (void)out_size; (void)ws_size;

    const float* kn_emb  = (const float*)d_in[0];
    const float* exer    = (const float*)d_in[1];
    const float* stu     = (const float*)d_in[2];
    const float* fc[6]   = { (const float*)d_in[3], (const float*)d_in[5], (const float*)d_in[7],
                             (const float*)d_in[9], (const float*)d_in[11], (const float*)d_in[13] };
    const float* attn[6] = { (const float*)d_in[4], (const float*)d_in[6], (const float*)d_in[8],
                             (const float*)d_in[10], (const float*)d_in[12], (const float*)d_in[14] };
    const float* k1w = (const float*)d_in[15]; const float* k1b = (const float*)d_in[16];
    const float* k2w = (const float*)d_in[17]; const float* k2b = (const float*)d_in[18];
    const float* k3w = (const float*)d_in[19]; const float* k3b = (const float*)d_in[20];
    const float* e1w = (const float*)d_in[21]; const float* e1b = (const float*)d_in[22];
    const float* e2w = (const float*)d_in[23]; const float* e2b = (const float*)d_in[24];
    const int* esrc[6] = { (const int*)d_in[25], (const int*)d_in[27], (const int*)d_in[29],
                           (const int*)d_in[31], (const int*)d_in[33], (const int*)d_in[35] };
    const int* edst[6] = { (const int*)d_in[26], (const int*)d_in[28], (const int*)d_in[30],
                           (const int*)d_in[32], (const int*)d_in[34], (const int*)d_in[36] };

    // ---- workspace carve-up ----
    char* base = (char*)d_ws;
    size_t off = 0;
    auto take = [&](size_t bytes) -> char* {
        char* p = base + off;
        off += (bytes + 255) & ~(size_t)255;
        return p;
    };
    const int NEU = NITEM + NUSER;   // 70000
    const int NEK = NITEM + NCPT;    // 20128
    _Float16* eu_h  = (_Float16*)take((size_t)NEU * DCH * 2);
    _Float16* ek_h  = (_Float16*)take((size_t)NEK * DCH * 2);
    _Float16* kn_h  = (_Float16*)take((size_t)NCPT * DCH * 2);
    _Float16* w16   = (_Float16*)take((size_t)6 * DCH * DCH * 2);
    float*    Z     = (float*)take((size_t)NEU * DCH * 4);
    float*    es_b  = (float*)take((size_t)NEU * 4);
    float*    ed_b  = (float*)take((size_t)NEU * 4);
    unsigned* menc  = (unsigned*)take((size_t)NEU * 4);
    float*    denom = (float*)take((size_t)NEU * 4);
    float*    ewb   = (float*)take((size_t)EUE_E * 4);
    float*    agg_dir = (float*)take((size_t)NCPT * DCH * 4);
    float*    agg_und = (float*)take((size_t)NCPT * DCH * 4);
    float*    agg_ke  = (float*)take((size_t)NCPT * DCH * 4);
    float*    agg_ek  = (float*)take((size_t)NITEM * DCH * 4);
    float*    agg_ue  = (float*)take((size_t)NUSER * DCH * 4);
    float*    agg_eu  = (float*)take((size_t)NITEM * DCH * 4);

    const int T = 256;
    auto blocks = [](long long n, int t) { return (unsigned)((n + t - 1) / t); };

    // ---- f16 tables: eu = [exer ; stu], ek = [exer ; kn], kn ----
    cvt_f16<<<blocks((long long)NITEM * DCH, T), T, 0, stream>>>(exer, eu_h, NITEM * DCH);
    cvt_f16<<<blocks((long long)NUSER * DCH, T), T, 0, stream>>>(stu, eu_h + (size_t)NITEM * DCH, NUSER * DCH);
    cvt_f16<<<blocks((long long)NITEM * DCH, T), T, 0, stream>>>(exer, ek_h, NITEM * DCH);
    cvt_f16<<<blocks((long long)NCPT * DCH, T), T, 0, stream>>>(kn_emb, ek_h + (size_t)NITEM * DCH, NCPT * DCH);
    cvt_f16<<<blocks((long long)NCPT * DCH, T), T, 0, stream>>>(kn_emb, kn_h, NCPT * DCH);
    for (int i = 0; i < 6; ++i)
        cvt_f16<<<blocks(DCH * DCH, T), T, 0, stream>>>(fc[i], w16 + (size_t)i * DCH * DCH, DCH * DCH);

    // ---- one GAT layer ----
    auto run_layer = [&](const _Float16* Htab, int nNodes, int li,
                         int nE, float* agg, int dstBase, int dstCount) {
        fill_u32<<<blocks(nNodes, T), T, 0, stream>>>(menc, 0u, nNodes);
        fill_u32<<<blocks(nNodes, T), T, 0, stream>>>((unsigned*)denom, 0u, nNodes);
        fill_u32<<<blocks((long long)dstCount * DCH, T), T, 0, stream>>>((unsigned*)agg, 0u, dstCount * DCH);
        wmma_gemm<<<(nNodes + 127) / 128, T, 0, stream>>>(Htab, w16 + (size_t)li * DCH * DCH, Z, nNodes);
        node_scalars<<<blocks(nNodes, 8), T, 0, stream>>>(Z, attn[li], es_b, ed_b, nNodes);
        edge_logits<<<blocks(nE, T), T, 0, stream>>>(esrc[li], edst[li], es_b, ed_b, ewb, menc, nE);
        edge_expw<<<blocks(nE, T), T, 0, stream>>>(edst[li], menc, ewb, denom, nE);
        edge_scatter<<<blocks((long long)nE * 32, T), T, 0, stream>>>(esrc[li], edst[li], ewb, denom,
                                                                      Z, agg, dstBase, dstCount, nE);
    };

    run_layer(kn_h, NCPT, 0, EK_E,  agg_dir, 0,     NCPT);   // dir
    run_layer(kn_h, NCPT, 1, EK_E,  agg_und, 0,     NCPT);   // und
    run_layer(ek_h, NEK,  2, EEK_E, agg_ke,  NITEM, NCPT);   // ke  (dst = concepts)
    run_layer(ek_h, NEK,  3, EEK_E, agg_ek,  0,     NITEM);  // ek  (dst = items)
    run_layer(eu_h, NEU,  4, EUE_E, agg_ue,  NITEM, NUSER);  // ue  (dst = users)
    run_layer(eu_h, NEU,  5, EUE_E, agg_eu,  0,     NITEM);  // eu  (dst = items)

    // ---- fusion heads -> d_out = [kn_out | exer_out | stu_out] ----
    float* out = (float*)d_out;
    float* out_kn = out;
    float* out_ex = out + (size_t)NCPT * DCH;
    float* out_st = out_ex + (size_t)NITEM * DCH;
    kn_fuse<<<blocks(NCPT, 8), T, 0, stream>>>(kn_emb, agg_dir, agg_und, agg_ke,
                                               k1w, k1b, k2w, k2b, k3w, k3b, out_kn);
    exer_fuse<<<blocks(NITEM, 8), T, 0, stream>>>(exer, agg_ek, agg_eu,
                                                  e1w, e1b, e2w, e2b, out_ex);
    stu_add<<<blocks((long long)NUSER * DCH, T), T, 0, stream>>>(stu, agg_ue, out_st,
                                                                 NUSER * DCH);
}
